// RecSSM_47141561041122
// MI455X (gfx1250) — matrix operand order repaired and checked by reading, not compiled, not run on hardware
//
#include <hip/hip_runtime.h>
#include <hip/hip_bf16.h>
#include <cstdint>

// ---------------------------------------------------------------------------
// Mamba2 vision model on gfx1250 (MI455X).
//   * GEMMs: bf16 v_wmma_f32_16x16x32_bf16, async global->LDS double-buffered
//     staging (global_load_async_to_lds_b128 + s_wait_asynccnt), 128x64 tile,
//     4 WMMA per wave per k-step.
//   * Weights kept in [N][K] layout == original tensor layout -> flat cast.
//   * SSM recurrence: fp32 VALU, wave32 per (batch, head), N=64 state in regs.
// ---------------------------------------------------------------------------

typedef __bf16 bf16_t;
typedef __attribute__((ext_vector_type(16))) __bf16 v16bf;
typedef __attribute__((ext_vector_type(8)))  __bf16 v8bf;
typedef __attribute__((ext_vector_type(8)))  float  v8f;

constexpr int E_   = 128;
constexpr int DI_  = 256;
constexpr int H_   = 8;
constexpr int N_   = 64;
constexpr int DC_  = 4;
constexpr int XBC_ = 384;   // DI + 2*N
constexpr int DIN_ = 648;   // 2*DI + 2*N + H
constexpr int NL_  = 4;
constexpr int B_   = 256;
constexpr int L_   = 256;
constexpr int TOK  = B_ * L_;   // 65536

__device__ __forceinline__ float silu_(float x) {
  return x / (1.f + __expf(-x));
}

// issue one 16B async global->LDS copy (per-lane addresses, ASYNCcnt-tracked)
__device__ __forceinline__ void async_cp16(const void* lds_ptr, const void* gptr) {
  unsigned lds_off = (unsigned)(uintptr_t)lds_ptr;
  asm volatile("global_load_async_to_lds_b128 %0, %1, off"
               :: "v"(lds_off), "v"(gptr) : "memory");
}
// constant-folded wait (callers pass literals; no branch emitted)
__device__ __forceinline__ void wait_async(int n) {
  if      (n == 0) asm volatile("s_wait_asynccnt 0x0" ::: "memory");
  else if (n == 1) asm volatile("s_wait_asynccnt 0x1" ::: "memory");
  else             asm volatile("s_wait_asynccnt 0x3" ::: "memory");
}

// ---------------------------------------------------------------------------
// Tiled bf16 WMMA GEMM:  C[M,N] = A[M,K] * BT[N,K]^T  (+bias) (+=C)
// A: bf16 row-major (or fused im2col gather from fp32 image x)
// BT: bf16 [N][K] row-major (original weight layout, cast to bf16)
// Block 256 thr (8 waves), tile 128x64, K-step 32, double-buffered async LDS.
// Wave -> 32x32 C strip (4 WMMA accumulators).
// ---------------------------------------------------------------------------
template<bool IM2COL, bool ACCUM>
__global__ __launch_bounds__(256)
void k_gemm(const bf16_t* __restrict__ A, const float* __restrict__ X,
            const bf16_t* __restrict__ BT, const float* __restrict__ bias,
            float* __restrict__ C, int M, int Nn, int K, int ldc)
{
  constexpr int BM = 128, BN = 64, BK = 32;
  constexpr int NEXT = IM2COL ? 1 : 3;     // asyncs in flight for next tile
  __shared__ bf16_t sA[2][BM][BK];   // [m][k]
  __shared__ bf16_t sB[2][BN][BK];   // [n][k]  (transposed tile: n-major)

  const int tid  = threadIdx.x;
  const int lane = tid & 31;
  const int wid  = tid >> 5;
  const int wm   = (wid & 3) * 32;   // wave row offset in tile
  const int wn   = (wid >> 2) * 32;  // wave col offset in tile
  const int m0   = blockIdx.x * BM;
  const int n0   = blockIdx.y * BN;
  const int half = lane >> 4;        // 0: lanes 0-15, 1: lanes 16-31
  const int l16  = lane & 15;
  const int klo  = half * 8;
  const int koff = half * 8;

  const int rA  = tid >> 1;          // 0..127 (A staging row)
  const int csA = (tid & 1) * 16;    // 0 or 16 (A staging k-segment, 16 elems)
  const int rB  = tid >> 2;          // 0..63  (B staging row)
  const int csB = (tid & 3) * 8;     // B staging k-segment (8 elems)
  const int ngc = min(n0 + rB, Nn - 1);   // clamp: uniform async issue count

  v8f c00 = {0.f,0.f,0.f,0.f,0.f,0.f,0.f,0.f};
  v8f c01 = c00, c10 = c00, c11 = c00;

  auto stage = [&](int buf, int k0) {
    if (IM2COL) {
      // token -> (b, ph, pw); k -> (c, p, q). 8-elem runs are contiguous in x.
      int row = m0 + rA;
      int b  = row >> 8;
      int ph = (row >> 4) & 15;
      int pw = row & 15;
#pragma unroll
      for (int s = 0; s < 2; ++s) {
        int k   = k0 + csA + s * 8;
        int c   = k >> 8;
        int rem = k & 255;
        int p   = rem >> 4;
        int q   = rem & 15;
        const float* xp = X + (((size_t)(b * 3 + c) * 256)
                               + (size_t)(ph * 16 + p)) * 256
                            + (size_t)(pw * 16 + q);
        float4 f0 = *(const float4*)(xp);
        float4 f1 = *(const float4*)(xp + 4);
        v8bf av;
        av[0] = (bf16_t)f0.x; av[1] = (bf16_t)f0.y;
        av[2] = (bf16_t)f0.z; av[3] = (bf16_t)f0.w;
        av[4] = (bf16_t)f1.x; av[5] = (bf16_t)f1.y;
        av[6] = (bf16_t)f1.z; av[7] = (bf16_t)f1.w;
        *(v8bf*)&sA[buf][rA][csA + s * 8] = av;
      }
    } else {
      const bf16_t* gA = A + (size_t)(m0 + rA) * K + k0 + csA;
      async_cp16(&sA[buf][rA][csA],     gA);
      async_cp16(&sA[buf][rA][csA + 8], gA + 8);
    }
    async_cp16(&sB[buf][rB][csB], BT + (size_t)ngc * K + k0 + csB);
  };

  auto compute = [&](int buf) {
    const bf16_t* arow0 = &sA[buf][wm + l16][0];
    const bf16_t* arow1 = &sA[buf][wm + 16 + l16][0];
    v8bf a0l = *(const v8bf*)(arow0 + klo);
    v8bf a0h = *(const v8bf*)(arow0 + klo + 16);
    v8bf a1l = *(const v8bf*)(arow1 + klo);
    v8bf a1h = *(const v8bf*)(arow1 + klo + 16);
    v16bf a0 = __builtin_shufflevector(a0l, a0h,
        0,1,2,3,4,5,6,7,8,9,10,11,12,13,14,15);
    v16bf a1 = __builtin_shufflevector(a1l, a1h,
        0,1,2,3,4,5,6,7,8,9,10,11,12,13,14,15);

    const bf16_t* b0p = &sB[buf][wn + l16][koff];
    const bf16_t* b1p = &sB[buf][wn + 16 + l16][koff];
    v8bf b0l = *(const v8bf*)(b0p);
    v8bf b0h = *(const v8bf*)(b0p + 16);
    v8bf b1l = *(const v8bf*)(b1p);
    v8bf b1h = *(const v8bf*)(b1p + 16);
    v16bf b0 = __builtin_shufflevector(b0l, b0h,
        0,1,2,3,4,5,6,7,8,9,10,11,12,13,14,15);
    v16bf b1 = __builtin_shufflevector(b1l, b1h,
        0,1,2,3,4,5,6,7,8,9,10,11,12,13,14,15);

    c00 = __builtin_amdgcn_wmma_f32_16x16x32_bf16(false, a0, false, b0,
                                                  (short)0, c00, false, false);
    c01 = __builtin_amdgcn_wmma_f32_16x16x32_bf16(false, a0, false, b1,
                                                  (short)0, c01, false, false);
    c10 = __builtin_amdgcn_wmma_f32_16x16x32_bf16(false, a1, false, b0,
                                                  (short)0, c10, false, false);
    c11 = __builtin_amdgcn_wmma_f32_16x16x32_bf16(false, a1, false, b1,
                                                  (short)0, c11, false, false);
  };

  // Double-buffered pipeline; last iteration peeled so both asynccnt waits
  // are compile-time immediates (no per-iteration branch around the asm).
  const int nk = K / BK;
  stage(0, 0);                                    // prologue: tile 0
  int k0 = 0;
  for (int i = 0; i < nk - 1; ++i, k0 += BK) {
    stage((i + 1) & 1, k0 + BK);                  // issue tile i+1
    wait_async(NEXT);                             // tile i resident (in-order)
    __syncthreads();
    compute(i & 1);
    __syncthreads();                              // all reads done before reuse
  }
  wait_async(0);                                  // last tile resident
  __syncthreads();
  compute((nk - 1) & 1);

  // ---- epilogue: D row = wm + 16*sub + half*8 + r, col = wn + (lane&15) ----
#pragma unroll
  for (int r = 0; r < 8; ++r) {
    int row0 = m0 + wm + half * 8 + r;
    int row1 = row0 + 16;
    int col0 = n0 + wn + l16;
    int col1 = col0 + 16;
    float b0v = bias ? bias[col0 < Nn ? col0 : 0] : 0.f;
    float b1v = bias ? bias[col1 < Nn ? col1 : 0] : 0.f;
    if (row0 < M) {
      if (col0 < Nn) {
        float v = c00[r] + b0v;
        if (ACCUM) v += C[(size_t)row0 * ldc + col0];
        C[(size_t)row0 * ldc + col0] = v;
      }
      if (col1 < Nn) {
        float v = c01[r] + b1v;
        if (ACCUM) v += C[(size_t)row0 * ldc + col1];
        C[(size_t)row0 * ldc + col1] = v;
      }
    }
    if (row1 < M) {
      if (col0 < Nn) {
        float v = c10[r] + b0v;
        if (ACCUM) v += C[(size_t)row1 * ldc + col0];
        C[(size_t)row1 * ldc + col0] = v;
      }
      if (col1 < Nn) {
        float v = c11[r] + b1v;
        if (ACCUM) v += C[(size_t)row1 * ldc + col1];
        C[(size_t)row1 * ldc + col1] = v;
      }
    }
  }
}

// ---------------------------------------------------------------------------
// LayerNorm over E=128 per token; OT=float (in-place ok) or bf16 (GEMM input)
// ---------------------------------------------------------------------------
template<typename OT>
__global__ __launch_bounds__(128)
void k_ln(const float* __restrict__ in, const float* __restrict__ w,
          const float* __restrict__ b, OT* __restrict__ out)
{
  __shared__ float s1[128], s2[128];
  int tok = blockIdx.x, d = threadIdx.x;
  float v = in[(size_t)tok * 128 + d];
  s1[d] = v; s2[d] = v * v;
  __syncthreads();
  for (int off = 64; off > 0; off >>= 1) {
    if (d < off) { s1[d] += s1[d + off]; s2[d] += s2[d + off]; }
    __syncthreads();
  }
  float mu  = s1[0] * (1.f / 128.f);
  float var = s2[0] * (1.f / 128.f) - mu * mu;
  float rs  = rsqrtf(var + 1e-5f);
  out[(size_t)tok * 128 + d] = (OT)((v - mu) * rs * w[d] + b[d]);
}

// softplus(dt + dt_bias) ; dts layout [tok][h]
__global__ void k_dt(const float* __restrict__ zx, const float* __restrict__ dtb,
                     float* __restrict__ dts)
{
  int i = blockIdx.x * blockDim.x + threadIdx.x;
  if (i >= TOK * H_) return;
  int h = i & 7, tok = i >> 3;
  float x = zx[(size_t)tok * DIN_ + (2 * DI_ + 2 * N_) + h] + dtb[h];
  dts[i] = (x > 20.f) ? x : log1pf(__expf(x));
}

// causal depthwise conv1d (DC=4) over L on xbc channels + SiLU
__global__ void k_conv_silu(const float* __restrict__ zx,
                            const float* __restrict__ cw,
                            const float* __restrict__ cb,
                            float* __restrict__ out)
{
  int idx = blockIdx.x * blockDim.x + threadIdx.x;
  if (idx >= TOK * XBC_) return;
  int ch = idx % XBC_;
  int l  = (idx / XBC_) % L_;
  int b  = idx / (XBC_ * L_);
  const float* src = zx + (size_t)b * L_ * DIN_ + DI_ + ch;   // (b, l', 256+ch)
  float acc = cb[ch];
#pragma unroll
  for (int k = 0; k < DC_; ++k) {
    int lp = l + k - (DC_ - 1);
    if (lp >= 0) acc += cw[ch * DC_ + k] * src[(size_t)lp * DIN_];
  }
  out[idx] = silu_(acc);
}

// ---------------------------------------------------------------------------
// SSM scan: one wave32 per (b,h); lane = p; h-state[N=64] in registers.
//   h = exp(dt*A)*h + (dt*x_p)*B_n ; y_p = sum_n h*C_n ; y += Dp*x_p
// B/C rows are 16B-aligned (row stride 1536B) -> float4 loads.
// ---------------------------------------------------------------------------
__global__ __launch_bounds__(256)
void k_scan(const float* __restrict__ xc, const float* __restrict__ dts,
            const float* __restrict__ A_log, const float* __restrict__ Dp,
            float* __restrict__ y)
{
  int gw   = (blockIdx.x * blockDim.x + threadIdx.x) >> 5;
  int lane = threadIdx.x & 31;
  int b = gw >> 3, h = gw & 7;

  float Ah = -__expf(A_log[h]);
  float Dh = Dp[h];

  float hs[N_];
#pragma unroll
  for (int n = 0; n < N_; ++n) hs[n] = 0.f;

  const float* xb  = xc  + (size_t)b * L_ * XBC_;
  const float* dtp = dts + (size_t)b * L_ * H_ + h;
  float*       yp  = y   + (size_t)b * L_ * DI_ + h * 32 + lane;

  for (int t = 0; t < L_; ++t) {
    float dtt = dtp[t * H_];
    float dA  = __expf(dtt * Ah);
    const float* row = xb + (size_t)t * XBC_;
    __builtin_prefetch(row + XBC_, 0, 1);   // next timestep row
    float xp   = row[h * 32 + lane];
    float coef = dtt * xp;
    float acc  = 0.f;
#pragma unroll
    for (int n4 = 0; n4 < N_ / 4; ++n4) {
      float4 B4 = *(const float4*)(row + DI_ + n4 * 4);
      float4 C4 = *(const float4*)(row + DI_ + N_ + n4 * 4);
      hs[n4*4+0] = dA * hs[n4*4+0] + coef * B4.x;  acc += hs[n4*4+0] * C4.x;
      hs[n4*4+1] = dA * hs[n4*4+1] + coef * B4.y;  acc += hs[n4*4+1] * C4.y;
      hs[n4*4+2] = dA * hs[n4*4+2] + coef * B4.z;  acc += hs[n4*4+2] * C4.z;
      hs[n4*4+3] = dA * hs[n4*4+3] + coef * B4.w;  acc += hs[n4*4+3] * C4.w;
    }
    yp[(size_t)t * DI_] = acc + Dh * xp;
  }
}

// y *= silu(z); RMS-norm over DI=256 with ssm_norm_w; emit bf16 for out-proj
__global__ __launch_bounds__(256)
void k_gate_norm(const float* __restrict__ y, const float* __restrict__ zx,
                 const float* __restrict__ nw, bf16_t* __restrict__ ybf)
{
  __shared__ float s[256];
  int tok = blockIdx.x, d = threadIdx.x;
  float z = zx[(size_t)tok * DIN_ + d];
  float v = y[(size_t)tok * DI_ + d] * silu_(z);
  s[d] = v * v;
  __syncthreads();
  for (int off = 128; off > 0; off >>= 1) {
    if (d < off) s[d] += s[d + off];
    __syncthreads();
  }
  float rs = rsqrtf(s[0] * (1.f / 256.f) + 1e-5f);
  ybf[(size_t)tok * DI_ + d] = (bf16_t)(v * rs * nw[d]);
}

// final LN over E then mean over L -> pooled[b][e]
__global__ __launch_bounds__(128)
void k_pool(const float* __restrict__ t, const float* __restrict__ w,
            const float* __restrict__ b, float* __restrict__ pooled)
{
  __shared__ float s1[128], s2[128];
  int bb = blockIdx.x, d = threadIdx.x;
  float acc = 0.f;
  for (int l = 0; l < L_; ++l) {
    float v = t[((size_t)bb * L_ + l) * 128 + d];
    s1[d] = v; s2[d] = v * v;
    __syncthreads();
    for (int off = 64; off > 0; off >>= 1) {
      if (d < off) { s1[d] += s1[d + off]; s2[d] += s2[d + off]; }
      __syncthreads();
    }
    float mu  = s1[0] * (1.f / 128.f);
    float var = s2[0] * (1.f / 128.f) - mu * mu;
    acc += (v - mu) * rsqrtf(var + 1e-5f) * w[d] + b[d];
    __syncthreads();
  }
  pooled[(size_t)bb * 128 + d] = acc * (1.f / 256.f);
}

// head GEMV + bias + L2-normalize (tiny: plain VALU)
__global__ __launch_bounds__(128)
void k_head(const float* __restrict__ pooled, const float* __restrict__ hw,
            const float* __restrict__ hb, float* __restrict__ out)
{
  __shared__ float s[128];
  int bb = blockIdx.x, o = threadIdx.x;
  const float* pv = pooled + (size_t)bb * 128;
  float acc = hb[o];
#pragma unroll 4
  for (int e = 0; e < 128; ++e) acc += pv[e] * hw[o * 128 + e];
  s[o] = acc * acc;
  __syncthreads();
  for (int off = 64; off > 0; off >>= 1) {
    if (o < off) s[o] += s[o + off];
    __syncthreads();
  }
  float nrm = fmaxf(sqrtf(s[0]), 1e-12f);
  out[(size_t)bb * 128 + o] = acc / nrm;
}

// flat fp32 -> bf16 cast (weights already sit in the [N][K] layout WMMA wants)
__global__ void k_cast(const float* __restrict__ src, bf16_t* __restrict__ dst,
                       int n)
{
  int i = blockIdx.x * blockDim.x + threadIdx.x;
  if (i < n) dst[i] = (bf16_t)src[i];
}

// ---------------------------------------------------------------------------
extern "C" void kernel_launch(void* const* d_in, const int* in_sizes, int n_in,
                              void* d_out, int out_size, void* d_ws, size_t ws_size,
                              hipStream_t stream)
{
  const float* x       = (const float*)d_in[0];
  const float* conv_w  = (const float*)d_in[1];
  const float* conv_b  = (const float*)d_in[2];
  const float* pe_ln_w = (const float*)d_in[3];
  const float* pe_ln_b = (const float*)d_in[4];
  const float* ln_w    = (const float*)d_in[5];
  const float* ln_b    = (const float*)d_in[6];
  const float* in_w    = (const float*)d_in[7];
  const float* c1w     = (const float*)d_in[8];
  const float* c1b     = (const float*)d_in[9];
  const float* dt_bias = (const float*)d_in[10];
  const float* A_log   = (const float*)d_in[11];
  const float* Dp      = (const float*)d_in[12];
  const float* nrm_w   = (const float*)d_in[13];
  const float* out_w   = (const float*)d_in[14];
  const float* fn_w    = (const float*)d_in[15];
  const float* fn_b    = (const float*)d_in[16];
  const float* head_w  = (const float*)d_in[17];
  const float* head_b  = (const float*)d_in[18];
  float* out = (float*)d_out;

  char* ws = (char*)d_ws;
  size_t off = 0;
  auto alloc = [&](size_t bytes) -> void* {
    void* p = ws + off;
    off = (off + bytes + 255) & ~(size_t)255;
    return p;
  };
  float*  t      = (float*) alloc((size_t)TOK * E_   * 4);
  bf16_t* tln    = (bf16_t*)alloc((size_t)TOK * E_   * 2);
  float*  zx     = (float*) alloc((size_t)TOK * DIN_ * 4);
  float*  xc     = (float*) alloc((size_t)TOK * XBC_ * 4);
  float*  yb     = (float*) alloc((size_t)TOK * DI_  * 4);
  bf16_t* ybf    = (bf16_t*)alloc((size_t)TOK * DI_  * 2);
  float*  dts    = (float*) alloc((size_t)TOK * H_   * 4);
  float*  pooled = (float*) alloc((size_t)B_  * E_   * 4);
  bf16_t* wBc    = (bf16_t*)alloc((size_t)128 * 768 * 2);          // [E][768]
  bf16_t* wBi    = (bf16_t*)alloc((size_t)NL_ * DIN_ * 128 * 2);   // [d][e]
  bf16_t* wBo    = (bf16_t*)alloc((size_t)NL_ * 128 * DI_ * 2);    // [e][d]

  // --- weight prep: flat casts (original layouts are already [N][K]) ---
  {
    int nc = 128 * 768;
    k_cast<<<(nc + 255) / 256, 256, 0, stream>>>(conv_w, wBc, nc);
    int ni = NL_ * DIN_ * 128;
    k_cast<<<(ni + 255) / 256, 256, 0, stream>>>(in_w, wBi, ni);
    int no = NL_ * 128 * DI_;
    k_cast<<<(no + 255) / 256, 256, 0, stream>>>(out_w, wBo, no);
  }

  // --- patch embed (fused im2col) + pe LayerNorm ---
  {
    dim3 g(TOK / 128, (128 + 63) / 64);   // N=128 -> 2 column tiles
    k_gemm<true, false><<<g, 256, 0, stream>>>(nullptr, x, wBc, conv_b,
                                               t, TOK, 128, 768, 128);
    k_ln<float><<<TOK, 128, 0, stream>>>(t, pe_ln_w, pe_ln_b, t);
  }

  // --- NL Mamba2 layers ---
  for (int i = 0; i < NL_; ++i) {
    k_ln<bf16_t><<<TOK, 128, 0, stream>>>(t, ln_w + i * E_, ln_b + i * E_, tln);

    dim3 gi(TOK / 128, (DIN_ + 63) / 64);
    k_gemm<false, false><<<gi, 256, 0, stream>>>(tln, nullptr,
        wBi + (size_t)i * DIN_ * 128, nullptr, zx, TOK, DIN_, 128, DIN_);

    int ndt = TOK * H_;
    k_dt<<<(ndt + 255) / 256, 256, 0, stream>>>(zx, dt_bias + i * H_, dts);

    int nc = TOK * XBC_;
    k_conv_silu<<<(nc + 255) / 256, 256, 0, stream>>>(zx, c1w + i * XBC_ * DC_,
                                                      c1b + i * XBC_, xc);

    k_scan<<<(B_ * H_ * 32) / 256, 256, 0, stream>>>(xc, dts, A_log + i * H_,
                                                     Dp + i * H_, yb);

    k_gate_norm<<<TOK, 256, 0, stream>>>(yb, zx, nrm_w + i * DI_, ybf);

    dim3 go(TOK / 128, (128 + 63) / 64);  // N=128 -> 2 column tiles
    k_gemm<false, true><<<go, 256, 0, stream>>>(ybf, nullptr,
        wBo + (size_t)i * 128 * DI_, nullptr, t, TOK, 128, DI_, 128);
  }

  // --- final LN + mean pool + head + L2 norm ---
  k_pool<<<B_, 128, 0, stream>>>(t, fn_w, fn_b, pooled);
  k_head<<<B_, 128, 0, stream>>>(pooled, head_w, head_b, out);
}
